// MultiHeadSelfAttention_2482491097268
// MI455X (gfx1250) — matrix operand order repaired and checked by reading, not compiled, not run on hardware
//
#include <hip/hip_runtime.h>
#include <hip/hip_bf16.h>

typedef __bf16 bf16_t;
typedef __attribute__((ext_vector_type(16))) __bf16 v16bf;
typedef __attribute__((ext_vector_type(8)))  float  v8f;

#define WMMA_BF16(a, b, c) \
  __builtin_amdgcn_wmma_f32_16x16x32_bf16(false, (a), false, (b), (short)0, (c), false, false)

// ---------------------------------------------------------------------------
// f32 -> bf16 conversion, 4 elems/thread (compiler merges the 4 b16 stores)
// ---------------------------------------------------------------------------
__global__ void cvt_f32_bf16(const float* __restrict__ in, bf16_t* __restrict__ out, int n) {
  int i = (blockIdx.x * blockDim.x + threadIdx.x) * 4;
  if (i < n) {
    float4 f = *(const float4*)(in + i);
    out[i + 0] = (bf16_t)f.x;
    out[i + 1] = (bf16_t)f.y;
    out[i + 2] = (bf16_t)f.z;
    out[i + 3] = (bf16_t)f.w;
  }
}

// ---------------------------------------------------------------------------
// Tiled bf16 WMMA GEMM: C[M,N] = A[M,K] @ B[K,N] + bias
//   OUT_MODE 0: bf16 head-major output [B, H=16, S, Hd=64]
//   OUT_MODE 1: f32 row-major output [M, N]
// Block tile 128x128, BK=32, 256 threads = 8 waves.
// Wave tile 32x64: wr = wave&3 -> rows 32*wr (2 A-frags), wc = wave>>2 ->
// cols 64*wc (4 B-frags) => 8 WMMAs per K-step from 6 fragment loads.
// Register double-buffering: next K-tile's global loads issued before the
// WMMA body; __builtin_prefetch warms L2 two steps ahead.
// ---------------------------------------------------------------------------
template <int OUT_MODE>
__global__ __launch_bounds__(256) void gemm_bf16_wmma(
    const bf16_t* __restrict__ A, const bf16_t* __restrict__ B,
    const float* __restrict__ bias, void* __restrict__ OutPtr,
    int M, int N, int K, int S) {
  constexpr int BM = 128, BN = 128, BK = 32, LDA = 40, LDB = 40;
  __shared__ bf16_t sA[BM * LDA];    // [row][k]
  __shared__ bf16_t sBt[BN * LDB];   // [n][k] (transposed)

  const int tid  = threadIdx.x;
  const int wave = tid >> 5;
  const int lane = tid & 31;
  const int m0 = blockIdx.y * BM;
  const int n0 = blockIdx.x * BN;
  const int wr = wave & 3;    // row tiles at 32*wr + {0,16}
  const int wc = wave >> 2;   // col tiles at 64*wc + 16*ct

  // staging indices: part p covers idx = tid + p*256 (0..511)
  int aRow[2], aGrp[2], bK[2], bGrp[2];
#pragma unroll
  for (int p = 0; p < 2; ++p) {
    int idx = tid + p * 256;
    aRow[p] = idx >> 2;  aGrp[p] = idx & 3;    // A: 128 rows x 4 groups of 8
    bK[p]   = idx >> 4;  bGrp[p] = idx & 15;   // B: 32 k x 16 groups of 8
  }

  uint4 ra[2], rb[2];
  auto loadTiles = [&](int ks) {
#pragma unroll
    for (int p = 0; p < 2; ++p) {
      ra[p] = *(const uint4*)(A + (size_t)(m0 + aRow[p]) * K + ks * BK + aGrp[p] * 8);
      rb[p] = *(const uint4*)(B + (size_t)(ks * BK + bK[p]) * N + n0 + bGrp[p] * 8);
    }
  };

  v8f acc[2][4] = {};
  const int r  = lane & 15;
  const int ka = (lane >> 4) * 16;
  const int nl = lane & 15;
  const int kb = (lane >> 4) * 16;

  const int kSteps = K / BK;
  loadTiles(0);

  for (int ks = 0; ks < kSteps; ++ks) {
    // ---- commit staged registers to LDS ----
#pragma unroll
    for (int p = 0; p < 2; ++p) {
      *(uint4*)&sA[aRow[p] * LDA + aGrp[p] * 8] = ra[p];
      const bf16_t* ds = (const bf16_t*)&rb[p];
#pragma unroll
      for (int j = 0; j < 8; ++j)
        sBt[(bGrp[p] * 8 + j) * LDB + bK[p]] = ds[j];
    }
    __syncthreads();

    // ---- issue next tile's global loads (latency hidden under WMMAs) ----
    if (ks + 1 < kSteps) loadTiles(ks + 1);
    if (ks + 2 < kSteps) {
      __builtin_prefetch(A + (size_t)(m0 + aRow[0]) * K + (ks + 2) * BK + aGrp[0] * 8, 0, 1);
      __builtin_prefetch(B + (size_t)((ks + 2) * BK + bK[0]) * N + n0 + bGrp[0] * 8, 0, 1);
    }

    // ---- A fragments (16x32): lane l%16 = row, K = 16*(l/16)+{0..15} ----
    v16bf afrag[2];
#pragma unroll
    for (int at = 0; at < 2; ++at) {
      const bf16_t* ap = &sA[(wr * 32 + at * 16 + r) * LDA];
#pragma unroll
      for (int i = 0; i < 8; ++i) {
        afrag[at][i]     = ap[ka + i];
        afrag[at][8 + i] = ap[8 + ka + i];
      }
    }
    // ---- B fragments (32x16): lane l%16 = col, K = 16*(l/16)+{0..15} ----
#pragma unroll
    for (int ct = 0; ct < 4; ++ct) {
      v16bf bfrag;
      const bf16_t* bp = &sBt[(wc * 64 + ct * 16 + nl) * LDB + kb];
#pragma unroll
      for (int i = 0; i < 16; ++i) bfrag[i] = bp[i];
#pragma unroll
      for (int at = 0; at < 2; ++at)
        acc[at][ct] = WMMA_BF16(afrag[at], bfrag, acc[at][ct]);
    }
    __syncthreads();
  }

  // ---- epilogue: C layout VGPR v -> row v + 8*(lane/16), col lane%16 ----
  const int bb = (OUT_MODE == 0) ? (m0 / S) : 0;  // 128 | S: uniform per block
#pragma unroll
  for (int ct = 0; ct < 4; ++ct) {
    int gn = n0 + wc * 64 + ct * 16 + nl;
    float bv = bias[gn];
#pragma unroll
    for (int at = 0; at < 2; ++at) {
#pragma unroll
      for (int v = 0; v < 8; ++v) {
        int gm = m0 + wr * 32 + at * 16 + v + 8 * (lane >> 4);
        float val = acc[at][ct][v] + bv;
        if (OUT_MODE == 0) {
          bf16_t* Out = (bf16_t*)OutPtr;
          int s = gm - bb * S;
          int h = gn >> 6, hd = gn & 63;
          Out[(((size_t)(bb * 16 + h)) * S + s) * 64 + hd] = (bf16_t)val;
        } else {
          float* Out = (float*)OutPtr;
          Out[(size_t)gm * N + gn] = val;
        }
      }
    }
  }
}

// ---------------------------------------------------------------------------
// Flash attention over [BH=32][S][64] bf16 Q/K/V. grid = (B*H, S/64).
// 256 threads = 8 waves; wave w: q-subtile qt = w&3 (16 query rows),
// col half ch = w>>2 (two 16-wide hd tiles). Scores never leave LDS.
// K/V tiles register-double-buffered across kv steps.
// ---------------------------------------------------------------------------
__global__ __launch_bounds__(256) void flash_attn_wmma(
    const bf16_t* __restrict__ Qh, const bf16_t* __restrict__ Kh,
    const bf16_t* __restrict__ Vh, bf16_t* __restrict__ Ctx, int S) {
  constexpr int LQ = 72, LK = 72, LV = 72, LS = 68, LP = 72;
  __shared__ bf16_t sQ[64 * LQ];    // [q][hd]
  __shared__ bf16_t sK[64 * LK];    // [key][hd]
  __shared__ bf16_t sVt[64 * LV];   // [hd][key] (transposed)
  __shared__ float  sS[64 * LS];    // scores f32 [q][key]
  __shared__ bf16_t sP[64 * LP];    // probs bf16 [q][key]
  __shared__ float  mrow[64], lrow[64], arow[64];

  const int tid  = threadIdx.x;
  const int wave = tid >> 5;
  const int lane = tid & 31;
  const int bh = blockIdx.x;
  const int q0 = blockIdx.y * 64;
  const size_t base = (size_t)bh * S * 64;

  int lRow[2], lGrp[2];
#pragma unroll
  for (int p = 0; p < 2; ++p) {
    int idx = tid + p * 256;
    lRow[p] = idx >> 3;  lGrp[p] = idx & 7;   // 64 rows x 8 groups of 8
  }

  // stage Q tile (64x64)
#pragma unroll
  for (int p = 0; p < 2; ++p)
    *(uint4*)&sQ[lRow[p] * LQ + lGrp[p] * 8] =
        *(const uint4*)(Qh + base + (size_t)(q0 + lRow[p]) * 64 + lGrp[p] * 8);
  if (tid < 64) { mrow[tid] = -3.0e38f; lrow[tid] = 0.f; }
  __syncthreads();

  const int qt = wave & 3, ch = wave >> 2;
  const int r  = lane & 15;
  const int ka = (lane >> 4) * 16;
  const int nl = lane & 15;
  const int kb = (lane >> 4) * 16;

  v16bf qfrag[2];
#pragma unroll
  for (int kk = 0; kk < 2; ++kk) {
    const bf16_t* qp = &sQ[(qt * 16 + r) * LQ + kk * 32];
#pragma unroll
    for (int i = 0; i < 8; ++i) {
      qfrag[kk][i]     = qp[ka + i];
      qfrag[kk][8 + i] = qp[8 + ka + i];
    }
  }

  uint4 rk[2], rv[2];
  auto loadKV = [&](int j) {
#pragma unroll
    for (int p = 0; p < 2; ++p) {
      rk[p] = *(const uint4*)(Kh + base + (size_t)(j * 64 + lRow[p]) * 64 + lGrp[p] * 8);
      rv[p] = *(const uint4*)(Vh + base + (size_t)(j * 64 + lRow[p]) * 64 + lGrp[p] * 8);
    }
  };

  v8f o[2] = {};
  const float scale = 0.125f;  // 1/sqrt(64)

  const int nSteps = S / 64;
  loadKV(0);

  for (int j = 0; j < nSteps; ++j) {
    // ---- commit staged K (row-major) / V (transposed) to LDS ----
#pragma unroll
    for (int p = 0; p < 2; ++p) {
      *(uint4*)&sK[lRow[p] * LK + lGrp[p] * 8] = rk[p];
      const bf16_t* ds = (const bf16_t*)&rv[p];
#pragma unroll
      for (int e = 0; e < 8; ++e) sVt[(lGrp[p] * 8 + e) * LV + lRow[p]] = ds[e];
    }
    __syncthreads();
    if (j + 1 < nSteps) loadKV(j + 1);

    // ---- S = Q K^T: wave computes tiles (qt, kt), kt in {2ch, 2ch+1} ----
#pragma unroll
    for (int kt2 = 0; kt2 < 2; ++kt2) {
      int kt = ch * 2 + kt2;
      v8f sacc = {};
#pragma unroll
      for (int kk = 0; kk < 2; ++kk) {
        v16bf bfrag;
        const bf16_t* bp = &sK[(kt * 16 + nl) * LK + kk * 32 + kb];
#pragma unroll
        for (int i = 0; i < 16; ++i) bfrag[i] = bp[i];
        sacc = WMMA_BF16(qfrag[kk], bfrag, sacc);
      }
#pragma unroll
      for (int v = 0; v < 8; ++v)
        sS[(qt * 16 + v + 8 * (lane >> 4)) * LS + kt * 16 + nl] = sacc[v] * scale;
    }
    __syncthreads();

    // ---- online softmax: one row-owner thread per query row ----
    if (tid < 64) {
      const int row = tid;
      float m_old = mrow[row];
      float rmax = m_old;
#pragma unroll 8
      for (int c = 0; c < 64; ++c) rmax = fmaxf(rmax, sS[row * LS + c]);
      float alpha = __expf(m_old - rmax);
      float psum = 0.f;
#pragma unroll 8
      for (int c = 0; c < 64; ++c) {
        float p = __expf(sS[row * LS + c] - rmax);
        sP[row * LP + c] = (bf16_t)p;
        psum += p;
      }
      lrow[row] = alpha * lrow[row] + psum;
      mrow[row] = rmax;
      arow[row] = alpha;
    }
    __syncthreads();

    // ---- rescale O, then O += P @ V ----
    float al[8];
#pragma unroll
    for (int v = 0; v < 8; ++v) al[v] = arow[qt * 16 + v + 8 * (lane >> 4)];
#pragma unroll
    for (int t = 0; t < 2; ++t)
#pragma unroll
      for (int v = 0; v < 8; ++v) o[t][v] *= al[v];

    v16bf pfrag[2];
#pragma unroll
    for (int kk = 0; kk < 2; ++kk) {
      const bf16_t* pp = &sP[(qt * 16 + r) * LP + kk * 32];
#pragma unroll
      for (int i = 0; i < 8; ++i) {
        pfrag[kk][i]     = pp[ka + i];
        pfrag[kk][8 + i] = pp[8 + ka + i];
      }
    }
#pragma unroll
    for (int t = 0; t < 2; ++t) {
      int hdc = (ch * 2 + t) * 16 + nl;
#pragma unroll
      for (int kk = 0; kk < 2; ++kk) {
        v16bf vfrag;
        const bf16_t* vp = &sVt[hdc * LV + kk * 32 + kb];
#pragma unroll
        for (int i = 0; i < 16; ++i) vfrag[i] = vp[i];
        o[t] = WMMA_BF16(pfrag[kk], vfrag, o[t]);
      }
    }
    __syncthreads();
  }

  // ---- finalize: O /= l, write context bf16 as [B, S, D], D = h*64+hd ----
  float linv[8];
#pragma unroll
  for (int v = 0; v < 8; ++v) linv[v] = 1.0f / lrow[qt * 16 + v + 8 * (lane >> 4)];
  const int b = bh >> 4, h = bh & 15;
#pragma unroll
  for (int t = 0; t < 2; ++t) {
    int hdc = (ch * 2 + t) * 16 + nl;
#pragma unroll
    for (int v = 0; v < 8; ++v) {
      int s_idx = q0 + qt * 16 + v + 8 * (lane >> 4);
      Ctx[((size_t)b * S + s_idx) * 1024 + h * 64 + hdc] = (bf16_t)(o[t][v] * linv[v]);
    }
  }
}

// ---------------------------------------------------------------------------
extern "C" void kernel_launch(void* const* d_in, const int* in_sizes, int n_in,
                              void* d_out, int out_size, void* d_ws, size_t ws_size,
                              hipStream_t stream) {
  const float* x  = (const float*)d_in[0];
  const float* wq = (const float*)d_in[1];
  const float* bq = (const float*)d_in[2];
  const float* wk = (const float*)d_in[3];
  const float* bk = (const float*)d_in[4];
  const float* wv = (const float*)d_in[5];
  const float* bv = (const float*)d_in[6];
  const float* wo = (const float*)d_in[7];
  const float* bo = (const float*)d_in[8];

  const int D  = 1024;
  const int BS = in_sizes[0] / D;  // B*S = 4096
  const int B  = 2;
  const int S  = BS / B;           // 2048

  const size_t xE = (size_t)BS * D;   // 4M elems
  const size_t wE = (size_t)D * D;    // 1M elems

  char* ws = (char*)d_ws;
  bf16_t* xb   = (bf16_t*)ws;                      ws += xE * 2;
  bf16_t* wqb  = (bf16_t*)ws;                      ws += wE * 2;
  bf16_t* wkb  = (bf16_t*)ws;                      ws += wE * 2;
  bf16_t* wvb  = (bf16_t*)ws;                      ws += wE * 2;
  bf16_t* wob  = (bf16_t*)ws;                      ws += wE * 2;
  bf16_t* qb   = (bf16_t*)ws;                      ws += xE * 2;
  bf16_t* kbuf = (bf16_t*)ws;                      ws += xE * 2;
  bf16_t* vbuf = (bf16_t*)ws;                      ws += xE * 2;
  bf16_t* ctxb = (bf16_t*)ws;                      // xE * 2 more (48 MB total)

  // 1) precision conversion (bandwidth-trivial: ~40 MB at 23.3 TB/s)
  cvt_f32_bf16<<<(int)(xE / 1024), 256, 0, stream>>>(x,  xb,  (int)xE);
  cvt_f32_bf16<<<(int)(wE / 1024), 256, 0, stream>>>(wq, wqb, (int)wE);
  cvt_f32_bf16<<<(int)(wE / 1024), 256, 0, stream>>>(wk, wkb, (int)wE);
  cvt_f32_bf16<<<(int)(wE / 1024), 256, 0, stream>>>(wv, wvb, (int)wE);
  cvt_f32_bf16<<<(int)(wE / 1024), 256, 0, stream>>>(wo, wob, (int)wE);

  // 2) QKV projections (bf16 out, head-major layout)
  dim3 gg(D / 128, BS / 128);
  gemm_bf16_wmma<0><<<gg, 256, 0, stream>>>(xb, wqb, bq, qb,   BS, D, D, S);
  gemm_bf16_wmma<0><<<gg, 256, 0, stream>>>(xb, wkb, bk, kbuf, BS, D, D, S);
  gemm_bf16_wmma<0><<<gg, 256, 0, stream>>>(xb, wvb, bv, vbuf, BS, D, D, S);

  // 3) fused flash attention (avoids ~1 GB of f32 score traffic)
  flash_attn_wmma<<<dim3(B * 16, S / 64), 256, 0, stream>>>(qb, kbuf, vbuf, ctxb, S);

  // 4) output projection (f32 out + bias)
  gemm_bf16_wmma<1><<<gg, 256, 0, stream>>>(ctxb, wob, bo, d_out, BS, D, D, S);
}
